// RelativeSelfAttention_86096914416088
// MI455X (gfx1250) — compile-verified, matrix operand order checked by reading, not run on hardware
//
#include <hip/hip_runtime.h>
#include <hip/hip_bf16.h>
#include <math.h>

// ---------------------------------------------------------------------------
// Relative self-attention for MI455X (gfx1250, wave32, WMMA bf16 path).
// B=8, T=1024, C=512, H=8, HD=64, MAX_REL=68 (137 relative offsets).
//
// attn2 trick:  attn2[...,q,k] = Srel[...,q, clip(k-q)+68],  Srel = q @ rel_k^T
// w2 trick:     w2[q,:] = Pbin[q,:] @ rel_v,  Pbin = attention binned by offset
// x / weights / rel_k are pre-converted (weights transposed) to bf16 once so
// every WMMA A/B fragment is a single contiguous 32-byte load. rel_v is staged
// to LDS with global_load_async_to_lds_b128 (ASYNCcnt / s_wait_asynccnt).
// ---------------------------------------------------------------------------

typedef __bf16 bf16_t;
typedef __attribute__((ext_vector_type(16))) __bf16 v16bf;
typedef __attribute__((ext_vector_type(8)))  __bf16 v8bf;
typedef __attribute__((ext_vector_type(8)))  float  v8f;

#define B_      8
#define T_      1024
#define C_      512
#define H_      8
#define HD_     64
#define MAXREL  68
#define NREL    137   // 2*MAXREL+1
#define NRELP   144   // padded to 9x16 WMMA tiles
#define SCALE   0.04419417382415922f   // 1/sqrt(C)

// --------------------------- WMMA helper -----------------------------------
__device__ __forceinline__ v8f wmma_bf16(v16bf a, v16bf b, v8f c) {
    // D(16x16,f32) = A(16x32,bf16) * B(32x16,bf16) + C
    return __builtin_amdgcn_wmma_f32_16x16x32_bf16(
        /*neg_a=*/false, a, /*neg_b=*/false, b,
        /*c_mod=*/(short)0, c, /*reuse_a=*/false, /*reuse_b=*/false);
}

// Fragment (16x32) from a bf16 row-major source: one 32-byte vector load.
// A-frag: lane&15 = row, lane>>4 selects K 0..15 / 16..31 (contiguous k order, same as B).
// Also the B-frag (32x16) loader when the source is stored row-major by the
// WMMA N index (lane supplies column n; 16 contiguous elements = K-slice).
__device__ __forceinline__ v16bf frag_rm_bf16(const bf16_t* __restrict__ p0,
                                              int ld, int m0, int k0, int lane) {
    const bf16_t* p = p0 + (size_t)(m0 + (lane & 15)) * ld + k0 + ((lane >> 4) << 4);
    return *(const v16bf*)p;
}

// ------------- kernel 0a: transpose f32 weights -> bf16 Wt[n][k] ------------
__global__ __launch_bounds__(256) void wt_kernel(
    const float* __restrict__ Wq, const float* __restrict__ Wk,
    const float* __restrict__ Wv, const float* __restrict__ Wo,
    bf16_t* __restrict__ Wt)
{
    __shared__ float tile[32][33];
    const float* W = (blockIdx.z == 0) ? Wq : (blockIdx.z == 1) ? Wk
                   : (blockIdx.z == 2) ? Wv : Wo;
    bf16_t* out = Wt + (size_t)blockIdx.z * C_ * C_;
    const int k0 = blockIdx.x * 32, n0 = blockIdx.y * 32;
    const int tx = threadIdx.x & 31, ty = threadIdx.x >> 5;   // 32 x 8
#pragma unroll
    for (int r = ty; r < 32; r += 8) tile[r][tx] = W[(size_t)(k0 + r) * C_ + n0 + tx];
    __syncthreads();
#pragma unroll
    for (int r = ty; r < 32; r += 8)
        out[(size_t)(n0 + r) * C_ + k0 + tx] = (bf16_t)tile[tx][r];
}

// ------------- kernel 0b: x f32 -> bf16 (row-major copy) --------------------
__global__ __launch_bounds__(256) void xcvt_kernel(const float* __restrict__ src,
                                                   bf16_t* __restrict__ dst)
{
    const size_t i = ((size_t)blockIdx.x * 256 + threadIdx.x) * 8;
    v8bf r;
#pragma unroll
    for (int j = 0; j < 8; ++j) r[j] = (bf16_t)src[i + j];
    *(v8bf*)(dst + i) = r;
}

// ------------- kernel 0c: rel_k f32 -> bf16 ---------------------------------
__global__ __launch_bounds__(256) void rkcvt_kernel(const float* __restrict__ src,
                                                    bf16_t* __restrict__ dst, int n)
{
    const int i = blockIdx.x * 256 + threadIdx.x;
    if (i < n) dst[i] = (bf16_t)src[i];
}

// --------------------- kernel 1: fused QKV projection -----------------------
// y = x @ W + b for W in {Wq,Wk,Wv}. 32(M) x 64(N) per wave: 6 loads / 8 WMMAs
// per k-step. q,k stored bf16 [bh][t][hd]; v transposed bf16 [bh][hd][t].
__global__ __launch_bounds__(256) void qkv_gemm_kernel(
    const bf16_t* __restrict__ xb, const bf16_t* __restrict__ Wt,
    const float* __restrict__ bq, const float* __restrict__ bk,
    const float* __restrict__ bv,
    bf16_t* __restrict__ qb, bf16_t* __restrict__ kb, bf16_t* __restrict__ vT)
{
    const int lane = threadIdx.x & 31;
    const int gw   = blockIdx.x * 8 + (threadIdx.x >> 5);
    const int mt   = gw & 255;          // 256 M-tiles of 32 (M = B*T = 8192)
    const int rest = gw >> 8;
    const int n64  = rest & 7;          // 8 column groups of 64
    const int mat  = rest >> 3;         // 0=q 1=k 2=v
    const bf16_t* W   = Wt + (size_t)mat * C_ * C_;
    const float* bias = (mat == 0) ? bq : (mat == 1) ? bk : bv;

    const int m0 = mt * 32;
    v8f acc[2][4] = {};
#pragma unroll 1
    for (int k0 = 0; k0 < C_; k0 += 32) {
        if (k0 + 32 < C_)
            __builtin_prefetch(xb + (size_t)(m0 + (lane & 15)) * C_ + k0 + 32, 0, 3);
        v16bf a0 = frag_rm_bf16(xb, C_, m0,      k0, lane);
        v16bf a1 = frag_rm_bf16(xb, C_, m0 + 16, k0, lane);
#pragma unroll
        for (int s = 0; s < 4; ++s) {
            v16bf b = frag_rm_bf16(W, C_, n64 * 64 + s * 16, k0, lane);
            acc[0][s] = wmma_bf16(a0, b, acc[0][s]);
            acc[1][s] = wmma_bf16(a1, b, acc[1][s]);
        }
    }
    const int half = lane >> 4, l16 = lane & 15;
    const int bidx = m0 / T_;
    const int t0   = m0 % T_;
#pragma unroll
    for (int s = 0; s < 4; ++s) {
        const int n = n64 * 64 + s * 16 + l16;
        const int h = n >> 6, d = n & 63;
        const float bb = bias[n];
#pragma unroll
        for (int rg = 0; rg < 2; ++rg) {
#pragma unroll
            for (int vg = 0; vg < 8; ++vg) {
                const int t  = t0 + rg * 16 + vg + 8 * half;  // C-layout row
                const float v = acc[rg][s][vg] + bb;
                if (mat == 0)      qb[(size_t)((bidx * H_ + h) * T_ + t) * HD_ + d] = (bf16_t)v;
                else if (mat == 1) kb[(size_t)((bidx * H_ + h) * T_ + t) * HD_ + d] = (bf16_t)v;
                else               vT[(size_t)((bidx * H_ + h) * HD_ + d) * T_ + t] = (bf16_t)v;
            }
        }
    }
}

// ------------------- kernel 2: Srel = q @ rel_k^T ---------------------------
// Srel[bh][t][r] (r padded to 144), K = HD = 64 (two chained WMMAs).
__global__ __launch_bounds__(256) void srel_kernel(
    const bf16_t* __restrict__ qb, const bf16_t* __restrict__ rel_kb,
    float* __restrict__ Srel)
{
    const int lane = threadIdx.x & 31;
    const int gw   = blockIdx.x * 8 + (threadIdx.x >> 5);   // 4096 waves
    const int bh   = gw >> 6;
    const int q0   = (gw & 63) * 16;
    const int half = lane >> 4, l16 = lane & 15;

    const bf16_t* qp = qb + (size_t)bh * T_ * HD_;
    v16bf a0 = frag_rm_bf16(qp, HD_, q0, 0,  lane);
    v16bf a1 = frag_rm_bf16(qp, HD_, q0, 32, lane);
    float* srow = Srel + (size_t)bh * T_ * NRELP;

#pragma unroll 1
    for (int nt = 0; nt < 9; ++nt) {
        int r = nt * 16 + l16; if (r > NREL - 1) r = NREL - 1;   // clamp pad cols
        const bf16_t* p = rel_kb + (size_t)r * HD_ + (half << 4);
        v16bf b0 = *(const v16bf*)p;
        v16bf b1 = *(const v16bf*)(p + 32);
        v8f c = {};
        c = wmma_bf16(a0, b0, c);
        c = wmma_bf16(a1, b1, c);
#pragma unroll
        for (int vg = 0; vg < 8; ++vg)
            srow[(size_t)(q0 + vg + 8 * half) * NRELP + nt * 16 + l16] = c[vg];
    }
}

// --------------- kernel 3: flash attention with relative terms --------------
// Per wave: one 16-row q tile. Pass A: online row max + denominator.
// Pass B: normalized P -> LDS transpose -> WMMA P@V; P binned into Pbin (LDS
// atomics); epilogue adds Pbin @ rel_v (rel_v staged via async-to-LDS).
__global__ __launch_bounds__(256) void attn_kernel(
    const bf16_t* __restrict__ qb, const bf16_t* __restrict__ kb,
    const bf16_t* __restrict__ vT, const float* __restrict__ Srel,
    const float* __restrict__ rel_v, bf16_t* __restrict__ wb)
{
    __shared__ float s_pbin[8][16][NREL];                      // 70,144 B
    __shared__ __align__(16) float s_relv[9216];               // 36,864 B (pad)
    __shared__ __align__(32) bf16_t s_p[8][16][32];            //  8,192 B

    const int tid  = threadIdx.x;
    const int lane = tid & 31, wv = tid >> 5;
    const int half = lane >> 4, l16 = lane & 15;
    const int bh   = blockIdx.x >> 3;
    const int q0   = ((blockIdx.x & 7) * 8 + wv) * 16;

    // ---- stage rel_v f32 -> LDS with async DMA (ASYNCcnt-tracked) ----
    {
        const unsigned bytesTot = NREL * HD_ * 4u;             // 35,072
        const unsigned lbase = (unsigned)(uintptr_t)(void*)s_relv; // flat[31:0] == LDS offset
#pragma unroll 1
        for (int it = 0; it < 9; ++it) {
            unsigned off = (unsigned)(it * 256 + tid) * 16u;
            unsigned g   = (off <= bytesTot - 16u) ? off : (bytesTot - 16u);
            unsigned lds_addr = lbase + off;
            unsigned long long gaddr =
                (unsigned long long)(uintptr_t)((const char*)rel_v + g);
            asm volatile("global_load_async_to_lds_b128 %0, %1, off"
                         :: "v"(lds_addr), "v"(gaddr) : "memory");
        }
        asm volatile("s_wait_asynccnt 0x0" ::: "memory");
    }
    for (int i = lane; i < 16 * NREL; i += 32) ((float*)s_pbin[wv])[i] = 0.0f;
    __syncthreads();

    const bf16_t* qp = qb   + (size_t)bh * T_ * HD_;
    const bf16_t* kp = kb   + (size_t)bh * T_ * HD_;
    const bf16_t* vp = vT   + (size_t)bh * HD_ * T_;
    const float*  sp = Srel + (size_t)bh * T_ * NRELP;

    v16bf a0 = frag_rm_bf16(qp, HD_, q0, 0,  lane);
    v16bf a1 = frag_rm_bf16(qp, HD_, q0, 32, lane);

    float mrow[8], lrow[8];
#pragma unroll
    for (int vg = 0; vg < 8; ++vg) { mrow[vg] = -3.0e38f; lrow[vg] = 0.0f; }

    // ---------------- pass A: online max / denominator ----------------
#pragma unroll 1
    for (int kc = 0; kc < T_; kc += 32) {
        if (kc + 32 < T_)
            __builtin_prefetch(kp + (size_t)(kc + 32 + (lane & 15)) * HD_, 0, 3);
        float sv[2][8];
#pragma unroll
        for (int s = 0; s < 2; ++s) {
            v16bf b0 = frag_rm_bf16(kp, HD_, kc + s * 16, 0,  lane);
            v16bf b1 = frag_rm_bf16(kp, HD_, kc + s * 16, 32, lane);
            v8f c = {};
            c = wmma_bf16(a0, b0, c);
            c = wmma_bf16(a1, b1, c);
            const int kg = kc + s * 16 + l16;
#pragma unroll
            for (int vg = 0; vg < 8; ++vg) {
                const int qg = q0 + vg + 8 * half;
                int rd = kg - qg;
                rd = rd < -MAXREL ? -MAXREL : (rd > MAXREL ? MAXREL : rd);
                sv[s][vg] = (c[vg] + sp[(size_t)qg * NRELP + rd + MAXREL]) * SCALE;
            }
        }
#pragma unroll
        for (int vg = 0; vg < 8; ++vg) {
            float tm = fmaxf(sv[0][vg], sv[1][vg]);
#pragma unroll
            for (int m = 1; m < 16; m <<= 1) tm = fmaxf(tm, __shfl_xor(tm, m, 32));
            const float mn = fmaxf(mrow[vg], tm);
            float es = __expf(sv[0][vg] - mn) + __expf(sv[1][vg] - mn);
#pragma unroll
            for (int m = 1; m < 16; m <<= 1) es += __shfl_xor(es, m, 32);
            lrow[vg] = lrow[vg] * __expf(mrow[vg] - mn) + es;
            mrow[vg] = mn;
        }
    }
#pragma unroll
    for (int vg = 0; vg < 8; ++vg) lrow[vg] = 1.0f / lrow[vg];

    // ---------------- pass B: P@V and relative-V binning ----------------
    v8f o[4] = {};
#pragma unroll 1
    for (int kc = 0; kc < T_; kc += 32) {
#pragma unroll
        for (int s = 0; s < 2; ++s) {
            v16bf b0 = frag_rm_bf16(kp, HD_, kc + s * 16, 0,  lane);
            v16bf b1 = frag_rm_bf16(kp, HD_, kc + s * 16, 32, lane);
            v8f c = {};
            c = wmma_bf16(a0, b0, c);
            c = wmma_bf16(a1, b1, c);
            const int kg = kc + s * 16 + l16;
#pragma unroll
            for (int vg = 0; vg < 8; ++vg) {
                const int m  = vg + 8 * half;
                const int qg = q0 + m;
                int rd = kg - qg;
                const int rc = rd < -MAXREL ? 0 : (rd > MAXREL ? 2 * MAXREL : rd + MAXREL);
                const float pval =
                    __expf((c[vg] + sp[(size_t)qg * NRELP + rc]) * SCALE - mrow[vg]) * lrow[vg];
                s_p[wv][m][s * 16 + l16] = (bf16_t)pval;      // transpose via LDS
                atomicAdd(&s_pbin[wv][m][rc], pval);          // ds_add_f32
            }
        }
        // wave-internal LDS ordering before cross-lane fragment read
        asm volatile("s_wait_dscnt 0x0" ::: "memory");
        v16bf pa = *(const v16bf*)&s_p[wv][l16][half << 4];   // A-frag of P (16x32)
#pragma unroll
        for (int dt = 0; dt < 4; ++dt) {
            const bf16_t* p = vp + (size_t)(dt * 16 + l16) * T_ + kc + (half << 4);
            v16bf bvf = *(const v16bf*)p;                     // contiguous: v is transposed
            o[dt] = wmma_bf16(pa, bvf, o[dt]);
        }
    }

    // ---------------- epilogue: out = O + Pbin @ rel_v ----------------
    float w2[8][4];
#pragma unroll
    for (int vg = 0; vg < 8; ++vg)
#pragma unroll
        for (int dt = 0; dt < 4; ++dt) w2[vg][dt] = 0.0f;
#pragma unroll 1
    for (int r = 0; r < NREL; ++r) {
        float rv[4];
#pragma unroll
        for (int dt = 0; dt < 4; ++dt) rv[dt] = s_relv[r * HD_ + dt * 16 + l16];
#pragma unroll
        for (int vg = 0; vg < 8; ++vg) {
            const float pb = s_pbin[wv][vg + 8 * half][r];    // LDS broadcast in half
#pragma unroll
            for (int dt = 0; dt < 4; ++dt) w2[vg][dt] = fmaf(pb, rv[dt], w2[vg][dt]);
        }
    }
    bf16_t* wp = wb + (size_t)bh * T_ * HD_;
#pragma unroll
    for (int vg = 0; vg < 8; ++vg) {
        const int m = vg + 8 * half;
#pragma unroll
        for (int dt = 0; dt < 4; ++dt)
            wp[(size_t)(q0 + m) * HD_ + dt * 16 + l16] = (bf16_t)(o[dt][vg] + w2[vg][dt]);
    }
}

// ------------------- kernel 4: output projection ----------------------------
// out = concat_heads(w) @ Wo + bo ; head gather folded into the A-frag address.
// 32(M) x 64(N) per wave.
__global__ __launch_bounds__(256) void out_gemm_kernel(
    const bf16_t* __restrict__ wb, const bf16_t* __restrict__ WtO,
    const float* __restrict__ bo, float* __restrict__ out)
{
    const int lane = threadIdx.x & 31;
    const int gw   = blockIdx.x * 8 + (threadIdx.x >> 5);   // 2048 waves
    const int mt   = gw & 255;
    const int n64  = gw >> 8;                               // 0..7
    const int m0   = mt * 32;
    const int bidx = m0 / T_;
    const int t0   = m0 % T_;
    const int half = lane >> 4, l16 = lane & 15;

    v8f acc[2][4] = {};
#pragma unroll 1
    for (int k0 = 0; k0 < C_; k0 += 32) {
        const int koff = k0 + (half << 4);                  // stays within one head
        const size_t hb = (size_t)(bidx * H_ + (koff >> 6)) * T_;
        v16bf a0 = *(const v16bf*)(wb + (hb + t0 + l16)      * HD_ + (koff & 63));
        v16bf a1 = *(const v16bf*)(wb + (hb + t0 + 16 + l16) * HD_ + (koff & 63));
#pragma unroll
        for (int s = 0; s < 4; ++s) {
            v16bf b = frag_rm_bf16(WtO, C_, n64 * 64 + s * 16, k0, lane);
            acc[0][s] = wmma_bf16(a0, b, acc[0][s]);
            acc[1][s] = wmma_bf16(a1, b, acc[1][s]);
        }
    }
#pragma unroll
    for (int s = 0; s < 4; ++s) {
        const int n = n64 * 64 + s * 16 + l16;
        const float bb = bo[n];
#pragma unroll
        for (int rg = 0; rg < 2; ++rg)
#pragma unroll
            for (int vg = 0; vg < 8; ++vg)
                out[(size_t)(m0 + rg * 16 + vg + 8 * half) * C_ + n] = acc[rg][s][vg] + bb;
    }
}

// ---------------------------------------------------------------------------
extern "C" void kernel_launch(void* const* d_in, const int* in_sizes, int n_in,
                              void* d_out, int out_size, void* d_ws, size_t ws_size,
                              hipStream_t stream) {
    (void)in_sizes; (void)n_in; (void)out_size; (void)ws_size;
    const float* x     = (const float*)d_in[0];
    const float* Wq    = (const float*)d_in[1];
    const float* bq    = (const float*)d_in[2];
    const float* Wk    = (const float*)d_in[3];
    const float* bk    = (const float*)d_in[4];
    const float* Wv    = (const float*)d_in[5];
    const float* bv    = (const float*)d_in[6];
    const float* Wo    = (const float*)d_in[7];
    const float* bo    = (const float*)d_in[8];
    const float* rel_k = (const float*)d_in[9];
    const float* rel_v = (const float*)d_in[10];
    float* out = (float*)d_out;

    // workspace layout (needs ~79 MB)
    char* ws = (char*)d_ws;
    bf16_t* qb    = (bf16_t*)(ws);                        //  8 MiB  [bh][t][hd]
    bf16_t* kbuf  = (bf16_t*)(ws + (size_t)8  * 1048576); //  8 MiB  [bh][t][hd]
    bf16_t* vT    = (bf16_t*)(ws + (size_t)16 * 1048576); //  8 MiB  [bh][hd][t]
    bf16_t* wbuf  = (bf16_t*)(ws + (size_t)24 * 1048576); //  8 MiB  [bh][t][hd]
    float*  Srel  = (float*) (ws + (size_t)32 * 1048576); // 36 MiB  [bh][t][144]
    bf16_t* Wt    = (bf16_t*)(ws + (size_t)68 * 1048576); //  2 MiB  4x [n][k] bf16
    bf16_t* xb    = (bf16_t*)(ws + (size_t)70 * 1048576); //  8 MiB  [b*t][c] bf16
    bf16_t* relkb = (bf16_t*)(ws + (size_t)78 * 1048576); // 18 KiB  [137][64] bf16

    wt_kernel<<<dim3(16, 16, 4), 256, 0, stream>>>(Wq, Wk, Wv, Wo, Wt);
    xcvt_kernel<<<2048, 256, 0, stream>>>(x, xb);                 // 4M elems, 8/thread
    rkcvt_kernel<<<35, 256, 0, stream>>>(rel_k, relkb, NREL * HD_);
    qkv_gemm_kernel<<<768, 256, 0, stream>>>(xb, Wt, bq, bk, bv, qb, kbuf, vT);
    srel_kernel<<<512, 256, 0, stream>>>(qb, relkb, Srel);
    attn_kernel<<<512, 256, 0, stream>>>(qb, kbuf, vT, Srel, rel_v, wbuf);
    out_gemm_kernel<<<256, 256, 0, stream>>>(wbuf, Wt + (size_t)3 * C_ * C_, bo, out);
}
